// ResidualVQ2_9981503995964
// MI455X (gfx1250) — matrix operand order repaired and checked by reading, not compiled, not run on hardware
//
#include <hip/hip_runtime.h>
#include <hip/hip_bf16.h>

typedef __attribute__((ext_vector_type(16))) _Float16 v16h;
typedef __attribute__((ext_vector_type(8)))  float    v8f;
typedef __attribute__((ext_vector_type(4)))  unsigned u32x4;
typedef __attribute__((ext_vector_type(8)))  unsigned u32x8;

#define N_IN        131072
#define DIM         256
#define NUM_VQ      4
#define KCODES      512
#define BETA        0.2f

#define BLOCK       256
#define WAVES       8           // 256 threads / wave32
#define ROWS_PER_WAVE  16
#define ROWS_PER_BLOCK 128
#define GRID        (N_IN / ROWS_PER_BLOCK)   // 1024
#define CHUNK       128                        // codes per LDS tile
#define NCHUNK      (KCODES / CHUNK)           // 4
#define NTILE       (NUM_VQ * NCHUNK)          // 16 (stage,chunk) tiles
#define RSTRIDE     264         // fp32 residual row stride (8-float pad)
#define CSTRIDE     264         // f16 codebook row stride (8-half pad)

#define RES_BYTES   (ROWS_PER_BLOCK * RSTRIDE * 4)   // 135168
#define CB_BYTES    (CHUNK * CSTRIDE * 2)            // 67584

// TDM group1 word0: data_size | pad_enable | pad_interval | pad_amount
//  codebook tile: 2B elems, pad after 512B rows (64 x 8B -> code 6), pad 4 dwords (code 3)
#define TDM_W0_CB   ((1u << 16) | (1u << 20) | (6u << 22) | (3u << 25))
//  residual tile: 4B elems, pad after 1024B rows (128 x 8B -> code 7), pad 8 dwords (code 7)
#define TDM_W0_RES  ((2u << 16) | (1u << 20) | (7u << 22) | (7u << 25))

// workspace layout (bytes)
#define WS_CBH      0                                  // f16 codebooks: 1 MiB
#define WS_CBN      (NUM_VQ*KCODES*DIM*2)
#define WS_CNT      (WS_CBN + NUM_VQ*KCODES*4)
#define WS_SS       (WS_CNT + NUM_VQ*KCODES*4)

// output layout (float elements)
#define OUT_LOSS    ((size_t)N_IN * (NUM_VQ*DIM))
#define OUT_LOG     (OUT_LOSS + 1)
#define OUT_PPL     (OUT_LOG + N_IN)

// --------------------------------------------------------------- TDM 2D tile load
// One descriptor: count=1, type=2("image"); group1 packs dims/strides/pad.
// All operands are block-uniform -> stay in SGPRs for the "s" constraints.
__device__ __forceinline__ void tdm_load_2d(unsigned lds_off, unsigned long long gaddr,
                                            unsigned w0, unsigned tensor_d0,
                                            unsigned tensor_d1, unsigned tile_d0,
                                            unsigned tile_d1, unsigned stride0) {
  u32x4 g0;
  g0[0] = 1u;                                               // count=1, no gather
  g0[1] = lds_off;                                          // lds_addr (bytes)
  g0[2] = (unsigned)gaddr;                                  // global_addr[31:0]
  g0[3] = (unsigned)((gaddr >> 32) & 0x01FFFFFFull) | (2u << 30);  // addr[56:32] | type=2
  u32x8 g1;
  g1[0] = w0;                                               // mask=0 | data_size | pad cfg
  g1[1] = (tensor_d0 & 0xFFFFu) << 16;                      // tensor_dim0 lo16
  g1[2] = (tensor_d0 >> 16) | ((tensor_d1 & 0xFFFFu) << 16);// tensor_dim0 hi | dim1 lo
  g1[3] = (tensor_d1 >> 16) | (tile_d0 << 16);              // dim1 hi | tile_dim0
  g1[4] = tile_d1 & 0xFFFFu;                                // tile_dim1 (tile_dim2=0)
  g1[5] = stride0;                                          // tensor_dim0_stride lo32
  g1[6] = 0u;
  g1[7] = 0u;
  asm volatile("tensor_load_to_lds %0, %1" :: "s"(g0), "s"(g1) : "memory");
}

// ---------------------------------------------------------------- init: zero histogram
__global__ void rvq_init(unsigned* __restrict__ counts) {
  const int i = blockIdx.x * blockDim.x + threadIdx.x;
  if (i < NUM_VQ * KCODES) counts[i] = 0u;
}

// ------------------------------------------- prep: fp32 codebooks -> f16 copy + norms
__global__ void rvq_prep(const float* __restrict__ cb,
                         _Float16* __restrict__ cbh,
                         float* __restrict__ cbn) {
  const int code = blockIdx.x;        // 0 .. NUM_VQ*KCODES-1
  const int lane = threadIdx.x;       // 32 threads
  const float* src = cb + (size_t)code * DIM + lane * 8;
  const float4 a = *(const float4*)src;
  const float4 b = *(const float4*)(src + 4);
  _Float16* dst = cbh + (size_t)code * DIM + lane * 8;
  dst[0] = (_Float16)a.x; dst[1] = (_Float16)a.y;
  dst[2] = (_Float16)a.z; dst[3] = (_Float16)a.w;
  dst[4] = (_Float16)b.x; dst[5] = (_Float16)b.y;
  dst[6] = (_Float16)b.z; dst[7] = (_Float16)b.w;
  float s = a.x*a.x + a.y*a.y + a.z*a.z + a.w*a.w
          + b.x*b.x + b.y*b.y + b.z*b.z + b.w*b.w;
  #pragma unroll
  for (int m = 16; m >= 1; m >>= 1) s += __shfl_xor(s, m, 32);
  if (lane == 0) cbn[code] = s;
}

// ------------------------------------------------------------- fused 4-stage RVQ kernel
__global__ void __launch_bounds__(BLOCK)
rvq_main(const float* __restrict__ x, const float* __restrict__ cb,
         const int* __restrict__ emo, const _Float16* __restrict__ cbh,
         const float* __restrict__ cbn, unsigned* __restrict__ counts,
         float* __restrict__ sspart, float* __restrict__ out) {
  extern __shared__ char smem[];
  float* resLDS = (float*)smem;                            // 128 x RSTRIDE fp32

  const int tid  = threadIdx.x;
  const int wave = tid >> 5;
  const int lane = tid & 31;
  const int half = lane >> 4;
  const int lm   = lane & 15;
  const int block_row = blockIdx.x * ROWS_PER_BLOCK;

  // LDS byte offsets for TDM (generic-pointer low 32 bits = LDS offset)
  const unsigned res_off = (unsigned)(uintptr_t)smem;
  const unsigned cb0_off = res_off + RES_BYTES;

  // kick off TDM: residual tile + first codebook tile; wave 0 owns the DMA.
  if (wave == 0) {
    tdm_load_2d(res_off, (unsigned long long)(x + (size_t)block_row * DIM),
                TDM_W0_RES, DIM, ROWS_PER_BLOCK, DIM, ROWS_PER_BLOCK, DIM);
    tdm_load_2d(cb0_off, (unsigned long long)cbh,
                TDM_W0_CB, DIM, CHUNK, DIM, CHUNK, DIM);
    __builtin_amdgcn_s_wait_tensorcnt(0);
  }
  __syncthreads();

  float minv[8]; int mini[8];
  v16h Afrag[8];
  const float* rrow = resLDS + (wave * ROWS_PER_WAVE + lm) * RSTRIDE;

  for (int t = 0; t < NTILE; ++t) {
    const int stage = t >> 2;
    const int chunk = t & 3;

    // prefetch next (stage,chunk) codebook tile while we compute on this one
    if (wave == 0 && t + 1 < NTILE) {
      const size_t gbase = (size_t)(t + 1) * CHUNK * DIM;   // flat (stage*K + chunk*CHUNK)*DIM
      tdm_load_2d(cb0_off + (unsigned)(((t + 1) & 1) * CB_BYTES),
                  (unsigned long long)(cbh + gbase),
                  TDM_W0_CB, DIM, CHUNK, DIM, CHUNK, DIM);
    }

    if (chunk == 0) {
      #pragma unroll
      for (int j = 0; j < 8; ++j) { minv[j] = 3.4e38f; mini[j] = 0; }
      // rebuild A fragments from the (updated) residual: valid for all 4 chunks
      #pragma unroll
      for (int kc = 0; kc < 8; ++kc) {
        const float* p = rrow + kc * 32 + half * 16;
        const float4 fa = *(const float4*)(p);
        const float4 fb = *(const float4*)(p + 4);
        const float4 fc = *(const float4*)(p + 8);
        const float4 fd = *(const float4*)(p + 12);
        union { v16h v; _Float16 h[16]; } A;
        A.h[0]  = (_Float16)fa.x; A.h[1]  = (_Float16)fa.y;
        A.h[2]  = (_Float16)fa.z; A.h[3]  = (_Float16)fa.w;
        A.h[4]  = (_Float16)fb.x; A.h[5]  = (_Float16)fb.y;
        A.h[6]  = (_Float16)fb.z; A.h[7]  = (_Float16)fb.w;
        A.h[8]  = (_Float16)fc.x; A.h[9]  = (_Float16)fc.y;
        A.h[10] = (_Float16)fc.z; A.h[11] = (_Float16)fc.w;
        A.h[12] = (_Float16)fd.x; A.h[13] = (_Float16)fd.y;
        A.h[14] = (_Float16)fd.z; A.h[15] = (_Float16)fd.w;
        Afrag[kc] = A.v;
      }
    }

    // direct GEP from smem so address-space inference keeps these on the ds_ path
    const _Float16* cbuf = (const _Float16*)(smem + RES_BYTES + (t & 1) * CB_BYTES);

    #pragma unroll 2        // pipeline without blowing past 256 VGPRs
    for (int nt = 0; nt < CHUNK / 16; ++nt) {
      const int gcode = chunk * CHUNK + nt * 16 + lm;       // this lane's code id
      const float cn = cbn[stage * KCODES + gcode];         // issued before WMMA chain
      const _Float16* bp = cbuf + (nt * 16 + lm) * CSTRIDE + half * 16;
      union BU { v16h v; uint4 q[2]; } B[2];
      B[0].q[0] = *(const uint4*)(bp);
      B[0].q[1] = *(const uint4*)(bp + 8);
      v8f acc = {};
      #pragma unroll
      for (int kc = 0; kc < 8; ++kc) {
        if (kc < 7) {                                       // double-buffer B fragments
          B[(kc + 1) & 1].q[0] = *(const uint4*)(bp + (kc + 1) * 32);
          B[(kc + 1) & 1].q[1] = *(const uint4*)(bp + (kc + 1) * 32 + 8);
        }
        acc = __builtin_amdgcn_wmma_f32_16x16x32_f16(
            false, Afrag[kc], false, B[kc & 1].v, (short)0, acc, false, false);
      }
      #pragma unroll
      for (int j = 0; j < 8; ++j) {                         // rows M = half*8 + j
        const float d = fmaf(-2.0f, acc[j], cn);
        if (d < minv[j]) { minv[j] = d; mini[j] = gcode; }
      }
    }

    if (chunk == 3) {
      // min-reduce across the 16 lanes sharing each row
      #pragma unroll
      for (int j = 0; j < 8; ++j) {
        #pragma unroll
        for (int m = 8; m >= 1; m >>= 1) {
          const float ov = __shfl_xor(minv[j], m, 32);
          const int   oi = __shfl_xor(mini[j], m, 32);
          if (ov < minv[j] || (ov == minv[j] && oi < mini[j])) { minv[j] = ov; mini[j] = oi; }
        }
      }
      // gather winning codes (fp32), update residual, emit outputs
      float ss = 0.0f;
      const int d0 = lane * 8;
      #pragma unroll
      for (int r = 0; r < ROWS_PER_WAVE; ++r) {
        const int wi = __shfl(mini[r & 7], (r < 8) ? 0 : 16, 32);
        const float* q = cb + ((size_t)stage * KCODES + wi) * DIM + d0;
        const float4 q0 = *(const float4*)q;
        const float4 q1 = *(const float4*)(q + 4);
        float* rp = resLDS + (wave * ROWS_PER_WAVE + r) * RSTRIDE + d0;
        float4 r0 = *(const float4*)rp;
        float4 r1 = *(const float4*)(rp + 4);
        r0.x -= q0.x; r0.y -= q0.y; r0.z -= q0.z; r0.w -= q0.w;
        r1.x -= q1.x; r1.y -= q1.y; r1.z -= q1.z; r1.w -= q1.w;
        ss += r0.x*r0.x + r0.y*r0.y + r0.z*r0.z + r0.w*r0.w
            + r1.x*r1.x + r1.y*r1.y + r1.z*r1.z + r1.w*r1.w;
        *(float4*)rp       = r0;
        *(float4*)(rp + 4) = r1;
        const size_t orow = (size_t)block_row + wave * ROWS_PER_WAVE + r;
        float* op = out + orow * (NUM_VQ * DIM) + stage * DIM + d0;
        *(float4*)op       = q0;
        *(float4*)(op + 4) = q1;
        if (lane == 0) {
          atomicAdd(&counts[stage * KCODES + wi], 1u);
          if (stage == 0) out[OUT_LOG + orow] = (float)emo[wi];
        }
      }
      #pragma unroll
      for (int m = 16; m >= 1; m >>= 1) ss += __shfl_xor(ss, m, 32);
      if (lane == 0)
        sspart[((size_t)blockIdx.x * WAVES + wave) * NUM_VQ + stage] = ss;
    }

    // next tile's DMA must be complete (and everyone done with buffers) before t+1
    if (wave == 0) __builtin_amdgcn_s_wait_tensorcnt(0);
    __syncthreads();
  }
}

// ----------------------------------------------- finalize: loss + perplexities (deterministic)
__global__ void rvq_final(const unsigned* __restrict__ counts,
                          const float* __restrict__ sspart,
                          float* __restrict__ out) {
  __shared__ float red[512];
  const int tid = threadIdx.x;      // 512 threads
  float s = 0.0f;
  for (int i = tid; i < GRID * WAVES * NUM_VQ; i += 512) s += sspart[i];
  red[tid] = s;
  __syncthreads();
  for (int m = 256; m > 0; m >>= 1) {
    if (tid < m) red[tid] += red[tid + m];
    __syncthreads();
  }
  if (tid == 0)
    out[OUT_LOSS] = red[0] * ((1.0f + BETA) / ((float)N_IN * (float)DIM));

  for (int stage = 0; stage < NUM_VQ; ++stage) {
    __syncthreads();
    const float e = (float)counts[stage * KCODES + tid] * (1.0f / (float)N_IN);
    red[tid] = e * logf(e + 1e-10f);
    __syncthreads();
    for (int m = 256; m > 0; m >>= 1) {
      if (tid < m) red[tid] += red[tid + m];
      __syncthreads();
    }
    if (tid == 0) out[OUT_PPL + stage] = expf(-red[0]);
  }
}

// --------------------------------------------------------------------------- launcher
extern "C" void kernel_launch(void* const* d_in, const int* in_sizes, int n_in,
                              void* d_out, int out_size, void* d_ws, size_t ws_size,
                              hipStream_t stream) {
  const float* x   = (const float*)d_in[0];
  const float* cb  = (const float*)d_in[1];
  const int*   emo = (const int*)d_in[2];
  float* out = (float*)d_out;
  char*  ws  = (char*)d_ws;

  _Float16* cbh   = (_Float16*)(ws + WS_CBH);
  float*    cbn   = (float*)(ws + WS_CBN);
  unsigned* cnt   = (unsigned*)(ws + WS_CNT);
  float*    sspar = (float*)(ws + WS_SS);

  rvq_init<<<(NUM_VQ * KCODES + 255) / 256, 256, 0, stream>>>(cnt);
  rvq_prep<<<NUM_VQ * KCODES, 32, 0, stream>>>(cb, cbh, cbn);

  const size_t lds = (size_t)RES_BYTES + 2 * (size_t)CB_BYTES;   // ~264 KiB of 320 KiB
  rvq_main<<<GRID, BLOCK, lds, stream>>>(x, cb, emo, cbh, cbn, cnt, sspar, out);

  rvq_final<<<1, 512, 0, stream>>>(cnt, sspar, out);
}
